// QwenMoeTransformerDecoder_20813411516428
// MI455X (gfx1250) — compile-verified, matrix operand order checked by reading, not run on hardware
//
#include <hip/hip_runtime.h>
#include <hip/hip_bf16.h>

typedef _Float16 f16;
typedef __attribute__((ext_vector_type(8)))  _Float16 v8h;
typedef __attribute__((ext_vector_type(16))) _Float16 v16h;
typedef __attribute__((ext_vector_type(8)))  float    v8f;
typedef __attribute__((ext_vector_type(4)))  int      v4i;

// problem dims
#define BB   2
#define SS   2048
#define HH   2048
#define TT   (BB*SS)
#define EE   8
#define IM   1408
#define ISH  5632
#define NQH  16
#define NKV  2
#define DHD  128

// gemm epilogue flags
#define GF_BIAS      1
#define GF_SILU      2
#define GF_STORE16   4
#define GF_MUL16     8
#define GF_ROWSCALE 16
#define GF_ADDF32   32
#define GF_GATHER   64
#define GF_SCATTER 128

// opt-in async global->LDS pipeline (ASYNCcnt path, CDNA5 ISA)
// Probe-confirmed prototype (from clang diagnostic):
//   param0 = AS1 (global) v4i*  (non-const), param1 = AS3 (LDS) v4i*,
//   param2/3 = immediate int offset / cpol.
#if defined(__HIP_DEVICE_COMPILE__) && \
    __has_builtin(__builtin_amdgcn_global_load_async_to_lds_b128) && \
    __has_builtin(__builtin_amdgcn_s_wait_asynccnt)
#define GEMM_ASYNC 1
typedef __attribute__((address_space(1))) v4i glb_v4i;
typedef __attribute__((address_space(3))) v4i lds_v4i;
#define AS1P(p) ((glb_v4i*)(const void*)(p))
#define AS3P(p) ((lds_v4i*)(void*)(p))
#else
#define GEMM_ASYNC 0
#endif

// ---------------------------------------------------------------------------
// Fragment load: 16-bit A/B layout (CDNA5 ISA):
//   lane L: row = L%16, khalf = L/16
//   elements 0..7  -> K = khalf*8 + 0..7
//   elements 8..15 -> K = 16 + khalf*8 + 0..7
// caller passes p = rowBase + khalf*8 ; we read [p, p+8) and [p+16, p+24).
// ---------------------------------------------------------------------------
__device__ __forceinline__ v16h frag_load(const f16* p) {
    v8h lo = *(const v8h*)p;
    v8h hi = *(const v8h*)(p + 16);
    v16h r;
#pragma unroll
    for (int i = 0; i < 8; ++i) { r[i] = lo[i]; r[i + 8] = hi[i]; }
    return r;
}

__device__ __forceinline__ float rowmax16(float v) {
#pragma unroll
    for (int m = 1; m < 16; m <<= 1) v = fmaxf(v, __shfl_xor(v, m, 32));
    return v;
}
__device__ __forceinline__ float rowsum16(float v) {
#pragma unroll
    for (int m = 1; m < 16; m <<= 1) v += __shfl_xor(v, m, 32);
    return v;
}

// ---------------------------------------------------------------------------
// Weight convert + transpose: W f32 [K][N] -> Wt f16 [N][K]
// ---------------------------------------------------------------------------
__global__ void convtrans_kernel(const float* __restrict__ W, f16* __restrict__ Wt,
                                 int K, int N) {
    int n = blockIdx.x * 32 + threadIdx.x;
    int k = blockIdx.y * 8 + threadIdx.y;
    if (n < N && k < K) Wt[(size_t)n * K + k] = (f16)W[(size_t)k * N + n];
}

// ---------------------------------------------------------------------------
// RMSNorm (f32 in, f16 out)
// ---------------------------------------------------------------------------
__global__ __launch_bounds__(256) void rmsnorm_kernel(const float* __restrict__ x,
                                                      const float* __restrict__ scale,
                                                      f16* __restrict__ out, int H) {
    int t = blockIdx.x;
    const float* row = x + (size_t)t * H;
    float ss = 0.f;
    for (int i = threadIdx.x; i < H; i += 256) { float v = row[i]; ss += v * v; }
    __shared__ float red[256];
    red[threadIdx.x] = ss;
    __syncthreads();
    for (int s = 128; s > 0; s >>= 1) {
        if (threadIdx.x < s) red[threadIdx.x] += red[threadIdx.x + s];
        __syncthreads();
    }
    float r = rsqrtf(red[0] / (float)H + 1e-5f);
    f16* orow = out + (size_t)t * H;
    for (int i = threadIdx.x; i < H; i += 256) orow[i] = (f16)(row[i] * r * scale[i]);
}

// ---------------------------------------------------------------------------
// Generic WMMA GEMM.  A f16 [M][K], Bt f16 [N][K] (pre-transposed weights),
// C [M][N].  Block 256 thr = 8 waves; tile 64x256; wave tile 32x64; BK=32.
// Double-buffered LDS ping-pong: one barrier per K-step; async global->LDS
// (ASYNCcnt) when the toolchain exposes it, else register-staged loads.
// ---------------------------------------------------------------------------
template <int FLAGS>
__global__ __launch_bounds__(256) void gemm_wmma_kernel(
    const f16* __restrict__ A, const f16* __restrict__ Bt,
    const float* __restrict__ bias,
    float* __restrict__ Cf, f16* __restrict__ Ch,
    const float* __restrict__ addsrc, const f16* __restrict__ mulsrc,
    const float* __restrict__ rowscale, int rs_stride,
    const int* __restrict__ gidx, const int* __restrict__ gcnt,
    int M, int N, int K) {
    constexpr bool GATHER = (FLAGS & GF_GATHER) != 0;
    constexpr bool SCATTER = (FLAGS & GF_SCATTER) != 0;

    int Mrows = M;
    if constexpr (GATHER || SCATTER) { int c = *gcnt; Mrows = c < M ? c : M; }

    const int bm = blockIdx.y * 64;
    const int bn = blockIdx.x * 256;
    if (bm >= Mrows) return;

    __shared__ __align__(16) f16 As[2][64 * 48];
    __shared__ __align__(16) f16 Bs[2][256 * 48];

    const int tid = threadIdx.x;
    const int lane = tid & 31;
    const int wid = tid >> 5;
    const int wm = (wid >> 2) * 32;  // 0,32
    const int wn = (wid & 3) * 64;   // 0..192
    const int lrow = lane & 15;
    const int lk = lane >> 4;

    v8f acc[2][4];
#pragma unroll
    for (int i = 0; i < 2; ++i)
#pragma unroll
        for (int j = 0; j < 4; ++j)
#pragma unroll
            for (int e = 0; e < 8; ++e) acc[i][j][e] = 0.f;

    // global->LDS mapping
    const int arow_l = tid >> 2;        // 0..63
    const int akoff = (tid & 3) * 8;    // 0,8,16,24
    const int aGm = bm + arow_l;
    long asrc_row;
    if constexpr (GATHER) asrc_row = (aGm < Mrows) ? (long)gidx[aGm] : -1;
    else                  asrc_row = (aGm < Mrows) ? (long)aGm : -1;

    const int bGn = bn + tid;           // 0..255 rows of Bt
    const bool bvalid = bGn < N;

    const f16* aptr = (asrc_row >= 0) ? (A + (size_t)asrc_row * K + akoff) : nullptr;
    const f16* bptr = bvalid ? (Bt + (size_t)bGn * K) : nullptr;

    const int kTiles = K >> 5;

#if GEMM_ASYNC
    // DMA tiles straight into LDS; invalid rows left stale (their C rows/cols
    // are never stored, so garbage cannot escape).
    auto tile_issue = [&](int kt, int buf) {
        const int k0 = kt * 32;
        if (aptr)
            __builtin_amdgcn_global_load_async_to_lds_b128(
                AS1P(aptr + k0), AS3P(&As[buf][arow_l * 48 + akoff]), 0, 0);
        if (bptr) {
#pragma unroll
            for (int c = 0; c < 4; ++c)
                __builtin_amdgcn_global_load_async_to_lds_b128(
                    AS1P(bptr + k0 + c * 8), AS3P(&Bs[buf][tid * 48 + c * 8]), 0, 0);
        }
    };
    tile_issue(0, 0);
    __builtin_amdgcn_s_wait_asynccnt(0);
    __syncthreads();
#else
    v8h rA;
    v8h rB[4];
    auto tile_fetch = [&](int kt) {
        const int k0 = kt * 32;
        if (aptr) rA = *(const v8h*)(aptr + k0);
        else {
#pragma unroll
            for (int e = 0; e < 8; ++e) rA[e] = (f16)0.f;
        }
        if (bptr) {
#pragma unroll
            for (int c = 0; c < 4; ++c) rB[c] = *(const v8h*)(bptr + k0 + c * 8);
        } else {
#pragma unroll
            for (int c = 0; c < 4; ++c)
#pragma unroll
                for (int e = 0; e < 8; ++e) rB[c][e] = (f16)0.f;
        }
    };
    auto tile_commit = [&](int buf) {
        *(v8h*)(&As[buf][arow_l * 48 + akoff]) = rA;
#pragma unroll
        for (int c = 0; c < 4; ++c) *(v8h*)(&Bs[buf][tid * 48 + c * 8]) = rB[c];
    };
    tile_fetch(0);
    tile_commit(0);
    __syncthreads();
#endif

    for (int kt = 0; kt < kTiles; ++kt) {
        const int cur = kt & 1;
        const bool more = (kt + 1) < kTiles;
#if GEMM_ASYNC
        if (more) tile_issue(kt + 1, cur ^ 1);
#else
        if (more) tile_fetch(kt + 1);
#endif
        if (kt + 2 < kTiles) {  // global_prefetch_b8 into the hierarchy
            if (aptr) __builtin_prefetch(aptr + (kt + 2) * 32, 0, 0);
            if (bptr) __builtin_prefetch(bptr + (kt + 2) * 32, 0, 0);
        }

        v16h af[2], bf[4];
#pragma unroll
        for (int mi = 0; mi < 2; ++mi)
            af[mi] = frag_load(&As[cur][(wm + mi * 16 + lrow) * 48 + lk * 8]);
#pragma unroll
        for (int ni = 0; ni < 4; ++ni)
            bf[ni] = frag_load(&Bs[cur][(wn + ni * 16 + lrow) * 48 + lk * 8]);
#pragma unroll
        for (int mi = 0; mi < 2; ++mi)
#pragma unroll
            for (int ni = 0; ni < 4; ++ni)
                acc[mi][ni] = __builtin_amdgcn_wmma_f32_16x16x32_f16(
                    false, af[mi], false, bf[ni], (short)0, acc[mi][ni], false, false);

#if GEMM_ASYNC
        if (more) __builtin_amdgcn_s_wait_asynccnt(0);
#else
        if (more) tile_commit(cur ^ 1);
#endif
        __syncthreads();
    }

    // epilogue (C layout: elem i -> row = i + (lane/16)*8, col = lane%16)
#pragma unroll
    for (int mi = 0; mi < 2; ++mi)
#pragma unroll
        for (int ni = 0; ni < 4; ++ni)
#pragma unroll
            for (int i = 0; i < 8; ++i) {
                const int r = bm + wm + mi * 16 + i + lk * 8;
                const int c = bn + wn + ni * 16 + lrow;
                if (r < Mrows && c < N) {
                    float v = acc[mi][ni][i];
                    if constexpr ((FLAGS & GF_BIAS) != 0) v += bias[c];
                    if constexpr ((FLAGS & GF_SILU) != 0) v = v * (1.f / (1.f + expf(-v)));
                    if constexpr ((FLAGS & GF_MUL16) != 0) v *= (float)mulsrc[(size_t)r * N + c];
                    int srow = r;
                    if constexpr (SCATTER) srow = gidx[r];
                    if constexpr ((FLAGS & GF_ROWSCALE) != 0)
                        v *= rowscale[(size_t)srow * rs_stride];
                    if constexpr ((FLAGS & GF_ADDF32) != 0) v += addsrc[(size_t)r * N + c];
                    if constexpr (SCATTER) atomicAdd(&Cf[(size_t)srow * N + c], v);
                    else if constexpr ((FLAGS & GF_STORE16) != 0) Ch[(size_t)r * N + c] = (f16)v;
                    else Cf[(size_t)r * N + c] = v;
                }
            }
}

// ---------------------------------------------------------------------------
// RoPE (half-split) f32 -> f16 for q and k.  grid (T, NQH+NKV), block 64
// ---------------------------------------------------------------------------
__global__ __launch_bounds__(64) void rope_kernel(const float* __restrict__ qf,
                                                  const float* __restrict__ kf,
                                                  f16* __restrict__ q16,
                                                  f16* __restrict__ k16) {
    const int t = blockIdx.x;
    const int h = blockIdx.y;
    const int j = threadIdx.x;  // 0..63
    const int s = t % SS;
    const float inv = expf(-((float)(2 * j) / 128.f) * 9.210340371976184f); // ln(1e4)
    const float ang = (float)s * inv;
    const float c = cosf(ang), sn = sinf(ang);
    if (h < NQH) {
        const float* src = qf + (size_t)t * (NQH * DHD) + h * DHD;
        f16* dst = q16 + (size_t)t * (NQH * DHD) + h * DHD;
        float x1 = src[j], x2 = src[j + 64];
        dst[j] = (f16)(x1 * c - x2 * sn);
        dst[j + 64] = (f16)(x2 * c + x1 * sn);
    } else {
        const int hk = h - NQH;
        const float* src = kf + (size_t)t * (NKV * DHD) + hk * DHD;
        f16* dst = k16 + (size_t)t * (NKV * DHD) + hk * DHD;
        float x1 = src[j], x2 = src[j + 64];
        dst[j] = (f16)(x1 * c - x2 * sn);
        dst[j + 64] = (f16)(x2 * c + x1 * sn);
    }
}

// v f16 [B*S][NKV][128] -> vT f16 [B][NKV][128][S]
__global__ void vtrans_kernel(const f16* __restrict__ v, f16* __restrict__ vT) {
    const long total = (long)BB * NKV * DHD * SS;
    long i = (long)blockIdx.x * 256 + threadIdx.x;
    if (i >= total) return;
    int s = (int)(i % SS);
    int d = (int)((i / SS) % DHD);
    int kv = (int)((i / ((long)SS * DHD)) % NKV);
    int b = (int)(i / ((long)SS * DHD * NKV));
    vT[i] = v[((size_t)(b * SS + s) * NKV + kv) * DHD + d];
}

// ---------------------------------------------------------------------------
// Flash attention, WMMA; one wave per (qtile=16 rows, head, batch).
// Streams 32-key tiles; QK^T and PV on v_wmma_f32_16x16x32_f16.
// ---------------------------------------------------------------------------
__global__ __launch_bounds__(32) void flash_attn_kernel(const f16* __restrict__ q16,
                                                        const f16* __restrict__ k16,
                                                        const f16* __restrict__ vT,
                                                        f16* __restrict__ attn) {
    const int qt = blockIdx.x, h = blockIdx.y, b = blockIdx.z;
    const int qbase = qt * 16;
    const int kv = h >> 3;  // NQ/NKV = 8
    const int lane = threadIdx.x;
    const int lrow = lane & 15;
    const int lk = lane >> 4;

    __shared__ __align__(16) f16 Ps[16 * 48];

    // Q fragments (A operand), dh split into 4 chunks of 32
    v16h qf[4];
    {
        const f16* qp = q16 + ((size_t)(b * SS + qbase + lrow) * NQH + h) * DHD;
#pragma unroll
        for (int c = 0; c < 4; ++c) qf[c] = frag_load(qp + c * 32 + lk * 8);
    }

    v8f o[8];
    float m[8], l[8];
#pragma unroll
    for (int n = 0; n < 8; ++n)
#pragma unroll
        for (int i = 0; i < 8; ++i) o[n][i] = 0.f;
#pragma unroll
    for (int i = 0; i < 8; ++i) { m[i] = -3e38f; l[i] = 0.f; }

    const float sc = 0.08838834764831845f;  // 1/sqrt(128)
    const int ktEnd = (qbase + 15) >> 5;    // S % 32 == 0 -> all key loads in-bounds

    for (int kt = 0; kt <= ktEnd; ++kt) {
        const int kb = kt * 32;
        if (kt < ktEnd) {  // prefetch next key tile (K rows and V columns)
            __builtin_prefetch(k16 + ((size_t)(b * SS + kb + 32 + lrow) * NKV + kv) * DHD, 0, 0);
            __builtin_prefetch(vT + ((size_t)(b * NKV + kv) * DHD + lrow) * SS + kb + 32, 0, 0);
        }
        v8f s0, s1;
#pragma unroll
        for (int i = 0; i < 8; ++i) { s0[i] = 0.f; s1[i] = 0.f; }

#pragma unroll
        for (int c = 0; c < 4; ++c) {
            const f16* kp0 = k16 + ((size_t)(b * SS + kb + lrow) * NKV + kv) * DHD + c * 32 + lk * 8;
            v16h kf0 = frag_load(kp0);
            s0 = __builtin_amdgcn_wmma_f32_16x16x32_f16(false, qf[c], false, kf0, (short)0, s0, false, false);
            const f16* kp1 = k16 + ((size_t)(b * SS + kb + 16 + lrow) * NKV + kv) * DHD + c * 32 + lk * 8;
            v16h kf1 = frag_load(kp1);
            s1 = __builtin_amdgcn_wmma_f32_16x16x32_f16(false, qf[c], false, kf1, (short)0, s1, false, false);
        }

        float alpha[8];
#pragma unroll
        for (int i = 0; i < 8; ++i) {
            const int qr = qbase + i + lk * 8;
            float a0 = s0[i] * sc, a1 = s1[i] * sc;
            if (kb + lrow > qr) a0 = -3e38f;
            if (kb + 16 + lrow > qr) a1 = -3e38f;
            const float rm = rowmax16(fmaxf(a0, a1));
            const float mn = fmaxf(m[i], rm);
            alpha[i] = expf(m[i] - mn);
            const float p0 = expf(a0 - mn);
            const float p1 = expf(a1 - mn);
            l[i] = l[i] * alpha[i] + rowsum16(p0 + p1);
            m[i] = mn;
            s0[i] = p0; s1[i] = p1;
        }
#pragma unroll
        for (int n = 0; n < 8; ++n)
#pragma unroll
            for (int i = 0; i < 8; ++i) o[n][i] *= alpha[i];

        // P (C layout) -> LDS row-major [16 q][32 keys] -> A fragment
#pragma unroll
        for (int i = 0; i < 8; ++i) {
            Ps[(i + lk * 8) * 48 + lrow] = (f16)s0[i];
            Ps[(i + lk * 8) * 48 + 16 + lrow] = (f16)s1[i];
        }
        __syncthreads();
        v16h pf = frag_load(&Ps[lrow * 48 + lk * 8]);
#pragma unroll
        for (int n = 0; n < 8; ++n) {
            const f16* vp = vT + ((size_t)(b * NKV + kv) * DHD + n * 16 + lrow) * SS + kb + lk * 8;
            v16h vf = frag_load(vp);
            o[n] = __builtin_amdgcn_wmma_f32_16x16x32_f16(false, pf, false, vf, (short)0, o[n], false, false);
        }
        __syncthreads();
    }

#pragma unroll
    for (int n = 0; n < 8; ++n)
#pragma unroll
        for (int i = 0; i < 8; ++i) {
            const int qr = qbase + i + lk * 8;
            const int d = n * 16 + lrow;
            attn[(size_t)(b * SS + qr) * (NQH * DHD) + h * DHD + d] = (f16)(o[n][i] / l[i]);
        }
}

// ---------------------------------------------------------------------------
// Router: softmax(8) -> top-4 -> normalize; build per-expert token lists.
// ---------------------------------------------------------------------------
__global__ void router_topk_kernel(const float* __restrict__ logits,
                                   float* __restrict__ dense_w,
                                   int* __restrict__ cnt, int* __restrict__ list) {
    int t = blockIdx.x * blockDim.x + threadIdx.x;
    if (t >= TT) return;
    float w[8];
    float mx = -3e38f;
#pragma unroll
    for (int e = 0; e < 8; ++e) { w[e] = logits[t * 8 + e]; mx = fmaxf(mx, w[e]); }
    float sum = 0.f;
#pragma unroll
    for (int e = 0; e < 8; ++e) { w[e] = expf(w[e] - mx); sum += w[e]; }
#pragma unroll
    for (int e = 0; e < 8; ++e) w[e] /= sum;
    bool used[8];
#pragma unroll
    for (int e = 0; e < 8; ++e) { used[e] = false; dense_w[t * 8 + e] = 0.f; }
    int sel[4]; float selw[4]; float tot = 0.f;
#pragma unroll
    for (int k = 0; k < 4; ++k) {
        int best = 0; float bw = -1.f;
#pragma unroll
        for (int e = 0; e < 8; ++e)
            if (!used[e] && w[e] > bw) { bw = w[e]; best = e; }
        used[best] = true; sel[k] = best; selw[k] = bw; tot += bw;
    }
#pragma unroll
    for (int k = 0; k < 4; ++k) {
        dense_w[t * 8 + sel[k]] = selw[k] / tot;
        int slot = atomicAdd(&cnt[sel[k]], 1);
        list[sel[k] * TT + slot] = t;
    }
}

// sigmoid(t16 @ sgw) per token
__global__ __launch_bounds__(256) void sgate_kernel(const f16* __restrict__ t16,
                                                    const float* __restrict__ w,
                                                    float* __restrict__ sg) {
    int t = blockIdx.x;
    float s = 0.f;
    for (int i = threadIdx.x; i < HH; i += 256) s += (float)t16[(size_t)t * HH + i] * w[i];
    __shared__ float red[256];
    red[threadIdx.x] = s;
    __syncthreads();
    for (int st = 128; st > 0; st >>= 1) {
        if (threadIdx.x < st) red[threadIdx.x] += red[threadIdx.x + st];
        __syncthreads();
    }
    if (threadIdx.x == 0) sg[t] = 1.f / (1.f + expf(-red[0]));
}

__global__ void combine_kernel(const float* __restrict__ x1, const float* __restrict__ moe,
                               const float* __restrict__ shd, float* __restrict__ out, long n) {
    long i = (long)blockIdx.x * 256 + threadIdx.x;
    if (i < n) out[i] = x1[i] + moe[i] + shd[i];
}

// ---------------------------------------------------------------------------
extern "C" void kernel_launch(void* const* d_in, const int* in_sizes, int n_in,
                              void* d_out, int out_size, void* d_ws, size_t ws_size,
                              hipStream_t stream) {
    (void)in_sizes; (void)n_in; (void)out_size; (void)ws_size;
    const float* x    = (const float*)d_in[0];
    const float* ln1  = (const float*)d_in[1];
    const float* wq   = (const float*)d_in[2];
    const float* bq   = (const float*)d_in[3];
    const float* wk   = (const float*)d_in[4];
    const float* bk   = (const float*)d_in[5];
    const float* wv   = (const float*)d_in[6];
    const float* bv   = (const float*)d_in[7];
    const float* wo   = (const float*)d_in[8];
    const float* ln2  = (const float*)d_in[9];
    const float* rw   = (const float*)d_in[10];
    const float* wgat = (const float*)d_in[11];
    const float* wupw = (const float*)d_in[12];
    const float* wdnw = (const float*)d_in[13];
    const float* swg  = (const float*)d_in[14];
    const float* swu  = (const float*)d_in[15];
    const float* swd  = (const float*)d_in[16];
    const float* sgw  = (const float*)d_in[17];
    float* out = (float*)d_out;

    char* p = (char*)d_ws;
    auto carve = [&](size_t bytes) -> char* {
        char* r = p;
        p += (bytes + 255) & ~(size_t)255;
        return r;
    };
    f16*   wq_t   = (f16*)carve((size_t)2048 * HH * 2);
    f16*   wk_t   = (f16*)carve((size_t)256 * HH * 2);
    f16*   wv_t   = (f16*)carve((size_t)256 * HH * 2);
    f16*   wo_t   = (f16*)carve((size_t)HH * 2048 * 2);
    f16*   rw_t   = (f16*)carve((size_t)8 * HH * 2);
    f16*   wg_t   = (f16*)carve((size_t)EE * IM * HH * 2);
    f16*   wu_t   = (f16*)carve((size_t)EE * IM * HH * 2);
    f16*   wd_t   = (f16*)carve((size_t)EE * HH * IM * 2);
    f16*   swg_t  = (f16*)carve((size_t)ISH * HH * 2);
    f16*   swu_t  = (f16*)carve((size_t)ISH * HH * 2);
    f16*   swd_t  = (f16*)carve((size_t)HH * ISH * 2);
    f16*   h1     = (f16*)carve((size_t)TT * HH * 2);
    float* qf     = (float*)carve((size_t)TT * 2048 * 4);
    float* kf     = (float*)carve((size_t)TT * 256 * 4);
    f16*   q16    = (f16*)carve((size_t)TT * 2048 * 2);
    f16*   k16    = (f16*)carve((size_t)TT * 256 * 2);
    f16*   v16    = (f16*)carve((size_t)TT * 256 * 2);
    f16*   vTb    = (f16*)carve((size_t)BB * NKV * DHD * SS * 2);
    f16*   attn16 = (f16*)carve((size_t)TT * 2048 * 2);
    float* x1     = (float*)carve((size_t)TT * HH * 4);
    f16*   t16    = (f16*)carve((size_t)TT * HH * 2);
    float* logits = (float*)carve((size_t)TT * 8 * 4);
    float* denw   = (float*)carve((size_t)TT * 8 * 4);
    int*   cnt    = (int*)carve(8 * 4);
    int*   list   = (int*)carve((size_t)EE * TT * 4);
    f16*   gbuf   = (f16*)carve((size_t)EE * TT * IM * 2);
    f16*   gu     = (f16*)carve((size_t)EE * TT * IM * 2);
    float* moe    = (float*)carve((size_t)TT * HH * 4);
    f16*   gsh    = (f16*)carve((size_t)TT * ISH * 2);
    f16*   ush    = (f16*)carve((size_t)TT * ISH * 2);
    float* sg     = (float*)carve((size_t)TT * 4);
    float* shd    = (float*)carve((size_t)TT * HH * 4);

    const dim3 ctb(32, 8);
    auto ctg = [](int K, int N) { return dim3((N + 31) / 32, (K + 7) / 8); };

    // --- weight convert/transpose (f32 [K][N] -> f16 [N][K]) ---
    convtrans_kernel<<<ctg(HH, 2048), ctb, 0, stream>>>(wq, wq_t, HH, 2048);
    convtrans_kernel<<<ctg(HH, 256), ctb, 0, stream>>>(wk, wk_t, HH, 256);
    convtrans_kernel<<<ctg(HH, 256), ctb, 0, stream>>>(wv, wv_t, HH, 256);
    convtrans_kernel<<<ctg(2048, HH), ctb, 0, stream>>>(wo, wo_t, 2048, HH);
    convtrans_kernel<<<ctg(HH, 8), ctb, 0, stream>>>(rw, rw_t, HH, 8);
    for (int e = 0; e < EE; ++e) {
        convtrans_kernel<<<ctg(HH, IM), ctb, 0, stream>>>(wgat + (size_t)e * HH * IM,
                                                          wg_t + (size_t)e * IM * HH, HH, IM);
        convtrans_kernel<<<ctg(HH, IM), ctb, 0, stream>>>(wupw + (size_t)e * HH * IM,
                                                          wu_t + (size_t)e * IM * HH, HH, IM);
        convtrans_kernel<<<ctg(IM, HH), ctb, 0, stream>>>(wdnw + (size_t)e * IM * HH,
                                                          wd_t + (size_t)e * HH * IM, IM, HH);
    }
    convtrans_kernel<<<ctg(HH, ISH), ctb, 0, stream>>>(swg, swg_t, HH, ISH);
    convtrans_kernel<<<ctg(HH, ISH), ctb, 0, stream>>>(swu, swu_t, HH, ISH);
    convtrans_kernel<<<ctg(ISH, HH), ctb, 0, stream>>>(swd, swd_t, ISH, HH);

    // --- attention block ---
    rmsnorm_kernel<<<TT, 256, 0, stream>>>(x, ln1, h1, HH);

    gemm_wmma_kernel<GF_BIAS><<<dim3(8, 64), 256, 0, stream>>>(
        h1, wq_t, bq, qf, nullptr, nullptr, nullptr, nullptr, 0, nullptr, nullptr, TT, 2048, HH);
    gemm_wmma_kernel<GF_BIAS><<<dim3(1, 64), 256, 0, stream>>>(
        h1, wk_t, bk, kf, nullptr, nullptr, nullptr, nullptr, 0, nullptr, nullptr, TT, 256, HH);
    gemm_wmma_kernel<GF_BIAS | GF_STORE16><<<dim3(1, 64), 256, 0, stream>>>(
        h1, wv_t, bv, nullptr, v16, nullptr, nullptr, nullptr, 0, nullptr, nullptr, TT, 256, HH);

    rope_kernel<<<dim3(TT, NQH + NKV), 64, 0, stream>>>(qf, kf, q16, k16);
    {
        long total = (long)BB * NKV * DHD * SS;
        vtrans_kernel<<<(int)((total + 255) / 256), 256, 0, stream>>>(v16, vTb);
    }
    flash_attn_kernel<<<dim3(SS / 16, NQH, BB), 32, 0, stream>>>(q16, k16, vTb, attn16);

    gemm_wmma_kernel<GF_ADDF32><<<dim3(8, 64), 256, 0, stream>>>(
        attn16, wo_t, nullptr, x1, nullptr, x, nullptr, nullptr, 0, nullptr, nullptr, TT, HH, 2048);

    // --- MoE block ---
    rmsnorm_kernel<<<TT, 256, 0, stream>>>(x1, ln2, t16, HH);
    gemm_wmma_kernel<0><<<dim3(1, 64), 256, 0, stream>>>(
        t16, rw_t, nullptr, logits, nullptr, nullptr, nullptr, nullptr, 0, nullptr, nullptr, TT, 8, HH);

    (void)hipMemsetAsync(cnt, 0, 8 * sizeof(int), stream);
    router_topk_kernel<<<TT / 256, 256, 0, stream>>>(logits, denw, cnt, list);
    (void)hipMemsetAsync(moe, 0, (size_t)TT * HH * sizeof(float), stream);

    for (int e = 0; e < EE; ++e) {
        const int* le = list + (size_t)e * TT;
        const int* ce = cnt + e;
        f16* gb = gbuf + (size_t)e * TT * IM;
        f16* gue = gu + (size_t)e * TT * IM;
        gemm_wmma_kernel<GF_GATHER | GF_SILU | GF_STORE16><<<dim3(6, 64), 256, 0, stream>>>(
            t16, wg_t + (size_t)e * IM * HH, nullptr, nullptr, gb, nullptr, nullptr, nullptr, 0,
            le, ce, TT, IM, HH);
        gemm_wmma_kernel<GF_GATHER | GF_MUL16 | GF_STORE16><<<dim3(6, 64), 256, 0, stream>>>(
            t16, wu_t + (size_t)e * IM * HH, nullptr, nullptr, gue, nullptr, gb, nullptr, 0,
            le, ce, TT, IM, HH);
        gemm_wmma_kernel<GF_SCATTER | GF_ROWSCALE><<<dim3(8, 64), 256, 0, stream>>>(
            gue, wd_t + (size_t)e * HH * IM, nullptr, moe, nullptr, nullptr, nullptr, denw + e, 8,
            le, ce, TT, HH, IM);
    }

    // --- shared expert ---
    gemm_wmma_kernel<GF_SILU | GF_STORE16><<<dim3(22, 64), 256, 0, stream>>>(
        t16, swg_t, nullptr, nullptr, gsh, nullptr, nullptr, nullptr, 0, nullptr, nullptr, TT, ISH, HH);
    gemm_wmma_kernel<GF_MUL16 | GF_STORE16><<<dim3(22, 64), 256, 0, stream>>>(
        t16, swu_t, nullptr, nullptr, ush, nullptr, gsh, nullptr, 0, nullptr, nullptr, TT, ISH, HH);
    sgate_kernel<<<TT, 256, 0, stream>>>(t16, sgw, sg);
    gemm_wmma_kernel<GF_ROWSCALE><<<dim3(8, 64), 256, 0, stream>>>(
        ush, swd_t, nullptr, shd, nullptr, nullptr, nullptr, sg, 1, nullptr, nullptr, TT, HH, ISH);

    // --- combine ---
    {
        long n = (long)TT * HH;
        combine_kernel<<<(int)((n + 255) / 256), 256, 0, stream>>>(x1, moe, shd, out, n);
    }
}
